// QuantizedLinear_85289460564390
// MI455X (gfx1250) — matrix operand order repaired and checked by reading, not compiled, not run on hardware
//
#include <hip/hip_runtime.h>

// ---------------------------------------------------------------------------
// QuantizedLinear + SVD outlier, MI455X (gfx1250, wave32, WMMA bf16 path)
//
//   out = X @ W_tot^T + bias,   W_tot[o][k] = Wq[o][k]*scale[o][k/32]
//                                           + sum_r U[o][r]*S[r]*V[k][r]
//
// Pass 1: fold_weights_kernel -> W_tot bf16 into d_ws (32 MB).
// Pass 1b (if ws allows): convert_x_kernel -> X bf16 (64 MB) into d_ws.
// Pass 2: qlinear_wmma_kernel -> bf16 WMMA GEMM, f32 accum, fused bias.
//         Tiles staged with GLOBAL_LOAD_ASYNC_TO_LDS_B128 + s_wait_asynccnt.
//         Steady-state loop is branchless (last stage peeled).
// ---------------------------------------------------------------------------

typedef __attribute__((ext_vector_type(16))) __bf16        v16bf;
typedef __attribute__((ext_vector_type(8)))  float         v8f;
typedef __attribute__((ext_vector_type(4)))  unsigned int  u32x4;
typedef __attribute__((ext_vector_type(2)))  unsigned int  u32x2;

#define TOK   8192
#define IN    4096
#define OUTN  4096
#define RANK  64
#define BLK   32            // mxfp8 block size along K
#define NBLK  (IN / BLK)    // 128

// f32 -> bf16 (RNE via hardware convert), packed pair
__device__ __forceinline__ unsigned pack_bf16x2(float x, float y) {
  unsigned short ua = __builtin_bit_cast(unsigned short, (__bf16)x);
  unsigned short ub = __builtin_bit_cast(unsigned short, (__bf16)y);
  return (unsigned)ua | ((unsigned)ub << 16);
}

// ---------------------------------------------------------------------------
// Pass 1: W_tot (bf16) = dequant(Wq) + U*diag(S)*V^T    (64x64 tile per WG)
// ---------------------------------------------------------------------------
__global__ __launch_bounds__(256) void fold_weights_kernel(
    const float* __restrict__ wq,
    const float* __restrict__ bscale,
    const float* __restrict__ svd_u,
    const float* __restrict__ svd_s,
    const float* __restrict__ svd_v,
    unsigned short* __restrict__ wb)
{
  __shared__ __align__(16) float Us[64 * 65];   // U[o][r]*S[r], padded
  __shared__ __align__(16) float Vs[64 * 65];   // V[k][r], padded
  const int tid = threadIdx.x;
  const int o0  = blockIdx.y * 64;
  const int k0  = blockIdx.x * 64;

#pragma unroll
  for (int i = 0; i < 4; ++i) {
    int c   = tid + i * 256;      // 1024 float4 chunks (64 rows x 16)
    int row = c >> 4;
    int q   = (c & 15) * 4;
    float4 uu  = *(const float4*)(svd_u + (size_t)(o0 + row) * RANK + q);
    float4 ssv = *(const float4*)(svd_s + q);
    Us[row * 65 + q + 0] = uu.x * ssv.x;
    Us[row * 65 + q + 1] = uu.y * ssv.y;
    Us[row * 65 + q + 2] = uu.z * ssv.z;
    Us[row * 65 + q + 3] = uu.w * ssv.w;
    float4 vv = *(const float4*)(svd_v + (size_t)(k0 + row) * RANK + q);
    Vs[row * 65 + q + 0] = vv.x;
    Vs[row * 65 + q + 1] = vv.y;
    Vs[row * 65 + q + 2] = vv.z;
    Vs[row * 65 + q + 3] = vv.w;
  }
  __syncthreads();

  const int oo = tid >> 2;          // 0..63 : output row within tile
  const int kk = (tid & 3) * 16;    // 0/16/32/48 : k column base (16 wide)

  float acc[16];
#pragma unroll
  for (int j = 0; j < 16; ++j) acc[j] = 0.f;

#pragma unroll 4
  for (int r = 0; r < RANK; ++r) {
    float uv = Us[oo * 65 + r];
#pragma unroll
    for (int j = 0; j < 16; ++j)
      acc[j] = fmaf(uv, Vs[(kk + j) * 65 + r], acc[j]);
  }

  const float sc = bscale[(size_t)(o0 + oo) * NBLK + (unsigned)(k0 + kk) / BLK];

  unsigned pk[8];
#pragma unroll
  for (int j4 = 0; j4 < 4; ++j4) {
    float4 w = *(const float4*)(wq + (size_t)(o0 + oo) * IN + (k0 + kk + j4 * 4));
    float r0 = fmaf(w.x, sc, acc[j4 * 4 + 0]);
    float r1 = fmaf(w.y, sc, acc[j4 * 4 + 1]);
    float r2 = fmaf(w.z, sc, acc[j4 * 4 + 2]);
    float r3 = fmaf(w.w, sc, acc[j4 * 4 + 3]);
    pk[j4 * 2 + 0] = pack_bf16x2(r0, r1);
    pk[j4 * 2 + 1] = pack_bf16x2(r2, r3);
  }
  unsigned short* dst = wb + (size_t)(o0 + oo) * IN + (k0 + kk);
  ((u32x4*)dst)[0] = (u32x4){pk[0], pk[1], pk[2], pk[3]};
  ((u32x4*)dst)[1] = (u32x4){pk[4], pk[5], pk[6], pk[7]};
}

// ---------------------------------------------------------------------------
// Pass 1b (optional): X f32 -> bf16, 8 elements per thread
// ---------------------------------------------------------------------------
__global__ __launch_bounds__(256) void convert_x_kernel(
    const float* __restrict__ x, unsigned short* __restrict__ xb)
{
  size_t i = ((size_t)blockIdx.x * 256 + threadIdx.x) * 8;
  float4 a = *(const float4*)(x + i);
  float4 b = *(const float4*)(x + i + 4);
  u32x4 p = {pack_bf16x2(a.x, a.y), pack_bf16x2(a.z, a.w),
             pack_bf16x2(b.x, b.y), pack_bf16x2(b.z, b.w)};
  *(u32x4*)(xb + i) = p;
}

// ---------------------------------------------------------------------------
// Pass 2: out[8192x4096] = bf16(X) @ Wb^T + bias, v_wmma_f32_16x16x32_bf16
// WG = 256 thr = 8 waves; tile 128x128, BK=64 (2 K-slices per stage);
// wave tile 64x32 (4x2 frags); 16 WMMAs between barriers.
// ---------------------------------------------------------------------------
#define BM  128
#define BN  128
#define BK  64
#define LDH 72   // LDS row stride in halves: 64 data + 8 pad = 144 B

__device__ __forceinline__ void async_copy_b128(unsigned lds_off,
                                                const void* gaddr) {
  // GLOBAL_LOAD_ASYNC_TO_LDS_B128: VDST = LDS byte offset, VADDR = 64b addr.
  asm volatile("global_load_async_to_lds_b128 %0, %1, off"
               :: "v"(lds_off), "v"(gaddr) : "memory");
}
__device__ __forceinline__ void wait_asynccnt0() {
  asm volatile("s_wait_asynccnt 0x0" ::: "memory");
}

template <bool PRECONV>
__global__ __launch_bounds__(256) void qlinear_wmma_kernel(
    const float* __restrict__ x,            // used when !PRECONV
    const unsigned short* __restrict__ xb,  // used when PRECONV
    const unsigned short* __restrict__ wb,
    const float* __restrict__ bias,
    float* __restrict__ out)
{
  __shared__ __align__(16) unsigned short As[2][BM * LDH];
  __shared__ __align__(16) unsigned short Bs[2][BN * LDH];

  const int tid  = threadIdx.x;
  const int lane = tid & 31;
  const int wave = tid >> 5;
  const int wm   = wave >> 2;      // 0..1 -> M offset of wave tile
  const int wn   = wave & 3;       // 0..3 -> N offset of wave tile
  const int l16  = lane & 15;
  const int hi   = lane >> 4;      // lane half selects K sub-range (ISA layout)
  const int m0   = blockIdx.y * BM;
  const int n0   = blockIdx.x * BN;

  float4 areg[8];   // f32 A staging (only when !PRECONV)

  // bf16 tile: 128 rows x 64 halves, 1024 x 16B chunks, 4 per thread (async)
  auto async_B = [&](int kt, int buf) {
    const int k0 = kt * BK;
#pragma unroll
    for (int i = 0; i < 4; ++i) {
      int c = tid + i * 256;
      int row = c >> 3, q = c & 7;
      async_copy_b128((unsigned)(size_t)&Bs[buf][row * LDH + q * 8],
                      wb + (size_t)(n0 + row) * IN + k0 + q * 8);
    }
  };
  auto async_A = [&](int kt, int buf) {
    const int k0 = kt * BK;
#pragma unroll
    for (int i = 0; i < 4; ++i) {
      int c = tid + i * 256;
      int row = c >> 3, q = c & 7;
      async_copy_b128((unsigned)(size_t)&As[buf][row * LDH + q * 8],
                      xb + (size_t)(m0 + row) * IN + k0 + q * 8);
    }
  };
  // A tile (f32 source): 128 rows x 64 f32, 8 float4 per thread
  auto load_A_f32 = [&](int kt) {
    const int k0 = kt * BK;
#pragma unroll
    for (int i = 0; i < 8; ++i) {
      int c = tid + i * 256;
      int row = c >> 4, c4 = c & 15;
      areg[i] = *(const float4*)(x + (size_t)(m0 + row) * IN + k0 + c4 * 4);
    }
  };
  auto store_A_lds = [&](int buf) {
#pragma unroll
    for (int i = 0; i < 8; ++i) {
      int c = tid + i * 256;
      int row = c >> 4, c4 = c & 15;
      u32x2 p = {pack_bf16x2(areg[i].x, areg[i].y),
                 pack_bf16x2(areg[i].z, areg[i].w)};
      *(u32x2*)(&As[buf][row * LDH + c4 * 4]) = p;
    }
  };

  union Frag { u32x4 q[2]; v16bf v; };

  v8f acc[4][2];
  const v8f vzero = {0.f, 0.f, 0.f, 0.f, 0.f, 0.f, 0.f, 0.f};
#pragma unroll
  for (int mf = 0; mf < 4; ++mf)
#pragma unroll
    for (int nf = 0; nf < 2; ++nf) acc[mf][nf] = vzero;

  auto compute = [&](int buf) {
    const unsigned short* ab = &As[buf][0];
    const unsigned short* bb = &Bs[buf][0];
#pragma unroll
    for (int ks = 0; ks < 2; ++ks) {          // two 16x16x32 K-slices
      // A frags: lanes<16 hold K{0..7,16..23}, lanes>=16 K{8..15,24..31}
      Frag a[4];
#pragma unroll
      for (int mf = 0; mf < 4; ++mf) {
        const unsigned short* p =
            ab + (wm * 64 + mf * 16 + l16) * LDH + ks * 32 + hi * 8;
        a[mf].q[0] = *(const u32x4*)(p);
        a[mf].q[1] = *(const u32x4*)(p + 16);
      }
      // B frags: lanes<16 hold K{0..7,16..23}, lanes>=16 K{8..15,24..31}
      Frag b[2];
#pragma unroll
      for (int nf = 0; nf < 2; ++nf) {
        const unsigned short* p =
            bb + (wn * 32 + nf * 16 + l16) * LDH + ks * 32 + hi * 8;
        b[nf].q[0] = *(const u32x4*)(p);
        b[nf].q[1] = *(const u32x4*)(p + 16);
      }
#pragma unroll
      for (int mf = 0; mf < 4; ++mf)
#pragma unroll
        for (int nf = 0; nf < 2; ++nf)
          acc[mf][nf] = __builtin_amdgcn_wmma_f32_16x16x32_bf16(
              false, a[mf].v, false, b[nf].v, (short)0, acc[mf][nf],
              false, false);
    }
  };

  // prologue: stage 0 in flight
  if constexpr (PRECONV) async_A(0, 0); else load_A_f32(0);
  async_B(0, 0);

  const int NK = IN / BK;   // 64 stages
  // steady state: branchless body, always prefetches next stage
  for (int kt = 0; kt < NK - 1; ++kt) {
    const int buf = kt & 1;
    if constexpr (!PRECONV) store_A_lds(buf);   // convert + ds_store
    wait_asynccnt0();                           // stage-kt async copies done
    __syncthreads();
    if constexpr (PRECONV) async_A(kt + 1, buf ^ 1);
    else load_A_f32(kt + 1);
    async_B(kt + 1, buf ^ 1);
    compute(buf);
  }
  // peeled last stage (no prefetch)
  {
    const int buf = (NK - 1) & 1;
    if constexpr (!PRECONV) store_A_lds(buf);
    wait_asynccnt0();
    __syncthreads();
    compute(buf);
  }

  // epilogue: +bias, store. C layout: vgpr j, lane l -> M = j + 8*(l>=16),
  // N = l%16.
#pragma unroll
  for (int nf = 0; nf < 2; ++nf) {
    const int col = n0 + wn * 32 + nf * 16 + l16;
    const float bv = bias[col];
#pragma unroll
    for (int mf = 0; mf < 4; ++mf) {
      const int rbase = m0 + wm * 64 + mf * 16 + hi * 8;
#pragma unroll
      for (int j = 0; j < 8; ++j)
        out[(size_t)(rbase + j) * OUTN + col] = acc[mf][nf][j] + bv;
    }
  }
}

// ---------------------------------------------------------------------------
extern "C" void kernel_launch(void* const* d_in, const int* in_sizes, int n_in,
                              void* d_out, int out_size, void* d_ws,
                              size_t ws_size, hipStream_t stream) {
  (void)in_sizes; (void)n_in; (void)out_size;
  const float* x     = (const float*)d_in[0];   // [8192,4096]
  const float* wq    = (const float*)d_in[1];   // [4096,4096]
  const float* bsc   = (const float*)d_in[2];   // [4096,128]
  const float* bias  = (const float*)d_in[3];   // [4096]
  const float* svd_u = (const float*)d_in[4];   // [4096,64]
  const float* svd_s = (const float*)d_in[5];   // [64]
  const float* svd_v = (const float*)d_in[6];   // [4096,64]
  float* out = (float*)d_out;

  const size_t WB_BYTES = (size_t)OUTN * IN * 2;     // 32 MB folded weight
  const size_t XB_BYTES = (size_t)TOK * IN * 2;      // 64 MB bf16 X
  unsigned short* wb = (unsigned short*)d_ws;
  unsigned short* xb = (unsigned short*)((char*)d_ws + WB_BYTES);

  fold_weights_kernel<<<dim3(IN / 64, OUTN / 64), 256, 0, stream>>>(
      wq, bsc, svd_u, svd_s, svd_v, wb);

  if (ws_size >= WB_BYTES + XB_BYTES) {
    // pre-convert X once: K-loop becomes pure async copy + ds_load + WMMA
    convert_x_kernel<<<(TOK * (IN / 8)) / 256, 256, 0, stream>>>(x, xb);
    qlinear_wmma_kernel<true><<<dim3(OUTN / BN, TOK / BM), 256, 0, stream>>>(
        x, xb, wb, bias, out);
  } else {
    qlinear_wmma_kernel<false><<<dim3(OUTN / BN, TOK / BM), 256, 0, stream>>>(
        x, xb, wb, bias, out);
  }
}
